// DWTBasedPMDBlock_15753940042176
// MI455X (gfx1250) — matrix operand order repaired and checked by reading, not compiled, not run on hardware
//
#include <hip/hip_runtime.h>
#include <hip/hip_bf16.h>

typedef __attribute__((ext_vector_type(16))) _Float16 v16h;
typedef __attribute__((ext_vector_type(8)))  float    v8f;

#define B_  16
#define C_  64
#define H_  256
#define W_  256

// ---------------------------------------------------------------------------
// Kernel 1: Haar DWT -> Perona-Malik damping (batch-0 details broadcast) ->
// inverse Haar DWT. Writes feat as f32 NCHW (residual) and f16 NHWC (conv in).
// Pure elementwise / bandwidth-bound.
// ---------------------------------------------------------------------------
__global__ void feat_kernel(const float* __restrict__ x,
                            float* __restrict__ feat32,
                            _Float16* __restrict__ feat16) {
  size_t idx = (size_t)blockIdx.x * blockDim.x + threadIdx.x;
  // one thread per (b,c,h2,w2): 16*64*128*128
  int w2 = (int)(idx & 127);
  size_t t = idx >> 7;
  int h2 = (int)(t & 127); t >>= 7;
  int c  = (int)(t & 63);
  int b  = (int)(t >> 6);
  if (b >= B_) return;

  size_t base = (((size_t)(b * C_ + c) * H_) + 2 * h2) * W_ + 2 * w2;
  float a  = x[base],       bb = x[base + 1];
  float cc = x[base + W_],  dd = x[base + W_ + 1];
  float LL = (a + bb + cc + dd) * 0.5f;
  float HH = (a - bb - cc + dd) * 0.5f;

  // detail bands of batch 0 at same (c,h2,w2) — reference broadcasts [:1]
  float LH0, HL0;
  if (b == 0) {
    LH0 = (a + bb - cc - dd) * 0.5f;
    HL0 = (a - bb + cc - dd) * 0.5f;
  } else {
    size_t b0 = (((size_t)c * H_) + 2 * h2) * W_ + 2 * w2;
    float a0 = x[b0],      b0v = x[b0 + 1];
    float c0 = x[b0 + W_], d0  = x[b0 + W_ + 1];
    LH0 = (a0 + b0v - c0 - d0) * 0.5f;
    HL0 = (a0 - b0v + c0 - d0) * 0.5f;
  }
  // g = 1/(1 + (sqrt(LH^2+HL^2)/K)^2) = 1/(1 + LH^2 + HL^2), K=1
  float g = 1.0f / (1.0f + LH0 * LH0 + HL0 * HL0);
  float dLH = g * LH0, dHL = g * HL0;

  float x00 = (LL + dLH + dHL + HH) * 0.5f;
  float x01 = (LL + dLH - dHL - HH) * 0.5f;
  float x10 = (LL - dLH + dHL - HH) * 0.5f;
  float x11 = (LL - dLH - dHL + HH) * 0.5f;

  feat32[base]          = x00;
  feat32[base + 1]      = x01;
  feat32[base + W_]     = x10;
  feat32[base + W_ + 1] = x11;

  // NHWC f16 copy: (((b*H + h)*W + w)*C + c)
  size_t n00 = ((((size_t)b * H_ + 2 * h2) * W_) + 2 * w2) * C_ + c;
  feat16[n00]               = (_Float16)x00;
  feat16[n00 + C_]          = (_Float16)x01;
  feat16[n00 + (size_t)W_ * C_]      = (_Float16)x10;
  feat16[n00 + (size_t)W_ * C_ + C_] = (_Float16)x11;
}

// ---------------------------------------------------------------------------
// Kernel 0: pack OIHW f32 3x3 weights into WMMA A-fragment layout (f16).
// K ordering: k = (kh*3+kw)*64 + ci  (tap-major, channel-minor).
// A 16x32 f16 order used here (contiguous k, B matches): lane 0-15 (M=lane): K {0..15};
// lane 16-31 (M=lane-16): K {16..31}.
// Fragment element index: ((mtile*18 + kstep)*32 + lane)*16 + j
// ---------------------------------------------------------------------------
__global__ void pack_weights_kernel(const float* __restrict__ w,
                                    _Float16* __restrict__ pw) {
  int i = blockIdx.x * blockDim.x + threadIdx.x;
  if (i >= 4 * 18 * 32 * 16) return;
  int j     = i & 15;
  int lane  = (i >> 4) & 31;
  int kstep = (i >> 9) % 18;
  int mtile = i / (16 * 32 * 18);
  int m  = mtile * 16 + (lane & 15);
  int kl = j + ((lane >= 16) ? 16 : 0);
  int k  = kstep * 32 + kl;
  int tap = k >> 6;      // k / 64
  int ci  = k & 63;      // k % 64
  int kh = tap / 3, kw = tap - kh * 3;
  pw[i] = (_Float16)w[(((size_t)m * C_ + ci) * 3 + kh) * 3 + kw];
}

// ---------------------------------------------------------------------------
// Implicit-GEMM conv3x3 (SAME) via v_wmma_f32_16x16x32_f16.
// Block tile: 64 output channels x 64 pixels of one row (b,h fixed).
// 256 threads = 8 wave32; wave -> 32-channel half x 16-pixel strip
// (2 accumulators). K loop: 9 taps x 2 channel-chunks = 18 WMMA steps.
// Input staged in LDS as [3 rows][66 cols][64 ch] f16 (channel-fastest) so
// every B fragment is one contiguous 32B v16h LDS read.
// Epilogue: conv1 -> BN+ReLU -> f16 NHWC; conv2 -> BN + residual -> f32 NCHW.
// ---------------------------------------------------------------------------
__global__ __launch_bounds__(256, 1) void conv3x3_wmma_kernel(
    const _Float16* __restrict__ in16,     // NHWC f16
    const _Float16* __restrict__ pw,       // packed A fragments
    const float* __restrict__ bias,
    const float* __restrict__ gamma, const float* __restrict__ beta,
    const float* __restrict__ mean,  const float* __restrict__ var,
    const float* __restrict__ residual32,  // f32 NCHW (conv2) or nullptr
    _Float16* __restrict__ out16,          // f16 NHWC (conv1) or nullptr
    float* __restrict__ out32,             // f32 NCHW (conv2) or nullptr
    int relu) {
  __shared__ __align__(16) _Float16 lds[3 * 66 * 64];

  int blk   = blockIdx.x;            // b*1024 + h*4 + wtile
  int wtile = blk & 3;
  int h     = (blk >> 2) & 255;
  int b     = blk >> 10;
  int w0    = wtile * 64;
  int tid   = threadIdx.x;

  // cooperative global->LDS stage, 16B vectors, zero-fill halo out of range
  for (int chunk = tid; chunk < (3 * 66 * 64) / 8; chunk += 256) {
    int off = chunk * 8;             // half index
    int row = off / (66 * 64);
    int rem = off - row * (66 * 64);
    int col = rem >> 6;
    int ci  = rem & 63;              // multiple of 8 -> stays in one (row,col)
    int hy  = h + row - 1;
    int wx  = w0 + col - 1;
    if (hy < 0 || hy >= H_ || wx < 0 || wx >= W_) {
      float4 z = {0.f, 0.f, 0.f, 0.f};
      *(float4*)&lds[off] = z;
    } else {
      const _Float16* src =
          in16 + ((((size_t)b * H_ + hy) * W_ + wx) * C_ + ci);
      *(float4*)&lds[off] = *(const float4*)src;
    }
  }
  __syncthreads();

  int wid    = tid >> 5;
  int lane   = tid & 31;
  int m_half = wid >> 2;          // 0..1 : which 32 output channels
  int n0     = (wid & 3) * 16;    // pixel strip within the 64-pixel tile
  int laneN  = lane & 15;
  int laneHi = lane >> 4;

  v8f acc0 = {}; v8f acc1 = {};
  const v16h* afrag = (const v16h*)pw;

  for (int tap = 0; tap < 9; ++tap) {
    int kh = tap / 3, kw = tap - kh * 3;
    for (int cch = 0; cch < 2; ++cch) {
      int kstep = tap * 2 + cch;
      // B fragment: lane -> pixel (n0+laneN), 16 contiguous channels
      int col = n0 + laneN + kw;                    // 0..65 (halo included)
      int cib = cch * 32 + laneHi * 16;
      v16h bfrag = *(const v16h*)&lds[(kh * 66 + col) * 64 + cib];
      // A fragments (L2-resident, broadcast across blocks)
      v16h a0 = afrag[((size_t)(m_half * 2 + 0) * 18 + kstep) * 32 + lane];
      v16h a1 = afrag[((size_t)(m_half * 2 + 1) * 18 + kstep) * 32 + lane];
      acc0 = __builtin_amdgcn_wmma_f32_16x16x32_f16(
          false, a0, false, bfrag, (short)0, acc0, false, false);
      acc1 = __builtin_amdgcn_wmma_f32_16x16x32_f16(
          false, a1, false, bfrag, (short)0, acc1, false, false);
    }
  }

  // Epilogue. C/D layout: VGPR r -> M = r + 8*laneHi, N = laneN.
  int w = w0 + n0 + laneN;
  for (int r = 0; r < 8; ++r) {
    for (int half = 0; half < 2; ++half) {
      float v  = (half == 0) ? acc0[r] : acc1[r];
      int   co = m_half * 32 + half * 16 + r + laneHi * 8;
      float inv = gamma[co] * rsqrtf(var[co] + 1e-5f);
      float sh  = beta[co] - mean[co] * inv + bias[co] * inv;
      float y   = v * inv + sh;
      if (relu) y = fmaxf(y, 0.0f);
      if (out16) {
        out16[(((size_t)b * H_ + h) * W_ + w) * C_ + co] = (_Float16)y;
      } else {
        size_t o = (((size_t)b * C_ + co) * H_ + h) * W_ + w;
        out32[o] = y + residual32[o];
      }
    }
  }
}

// ---------------------------------------------------------------------------
// Host-side launch. Input order: x,w1,b1,g1,be1,m1,v1,w2,b2,g2,be2,m2,v2.
// ---------------------------------------------------------------------------
extern "C" void kernel_launch(void* const* d_in, const int* in_sizes, int n_in,
                              void* d_out, int out_size, void* d_ws,
                              size_t ws_size, hipStream_t stream) {
  (void)in_sizes; (void)n_in; (void)out_size; (void)ws_size;
  const float* x   = (const float*)d_in[0];
  const float* w1  = (const float*)d_in[1];
  const float* b1  = (const float*)d_in[2];
  const float* g1  = (const float*)d_in[3];
  const float* be1 = (const float*)d_in[4];
  const float* m1  = (const float*)d_in[5];
  const float* v1  = (const float*)d_in[6];
  const float* w2  = (const float*)d_in[7];
  const float* b2  = (const float*)d_in[8];
  const float* g2  = (const float*)d_in[9];
  const float* be2 = (const float*)d_in[10];
  const float* m2  = (const float*)d_in[11];
  const float* v2  = (const float*)d_in[12];
  float* out = (float*)d_out;

  const size_t nElem   = (size_t)B_ * C_ * H_ * W_;      // 64M
  char* ws = (char*)d_ws;
  float*    feat32  = (float*)ws;                         // 256 MiB
  _Float16* feat16  = (_Float16*)(ws + nElem * 4);        // 128 MiB
  _Float16* mid16   = (_Float16*)(ws + nElem * 4 + nElem * 2);
  _Float16* pw1     = (_Float16*)(ws + nElem * 4 + nElem * 2 + nElem * 2);
  _Float16* pw2     = pw1 + 4 * 18 * 32 * 16;

  // pack weights into WMMA A-fragment layout
  pack_weights_kernel<<<144, 256, 0, stream>>>(w1, pw1);
  pack_weights_kernel<<<144, 256, 0, stream>>>(w2, pw2);

  // DWT -> PM damping -> iDWT (feat)
  {
    size_t threads = (size_t)B_ * C_ * (H_ / 2) * (W_ / 2); // 16,777,216
    feat_kernel<<<(unsigned)(threads / 256), 256, 0, stream>>>(x, feat32,
                                                               feat16);
  }

  const unsigned convBlocks = B_ * H_ * (W_ / 64);  // 16384

  // conv1 + BN + ReLU -> f16 NHWC
  conv3x3_wmma_kernel<<<convBlocks, 256, 0, stream>>>(
      feat16, pw1, b1, g1, be1, m1, v1,
      /*residual=*/nullptr, /*out16=*/mid16, /*out32=*/nullptr, /*relu=*/1);

  // conv2 + BN + residual add -> f32 NCHW output
  conv3x3_wmma_kernel<<<convBlocks, 256, 0, stream>>>(
      mid16, pw2, b2, g2, be2, m2, v2,
      /*residual=*/feat32, /*out16=*/nullptr, /*out32=*/out, /*relu=*/0);
}
